// PCD_Alignv2_26474178412705
// MI455X (gfx1250) — matrix operand order repaired and checked by reading, not compiled, not run on hardware
//
#include <hip/hip_runtime.h>

// ---------------------------------------------------------------------------
// Optional CDNA5 async global->LDS staging (ASYNCcnt path)
// ---------------------------------------------------------------------------
#if defined(__has_builtin)
#  if __has_builtin(__builtin_amdgcn_global_load_async_to_lds_b128) && \
      __has_builtin(__builtin_amdgcn_s_wait_asynccnt)
#    define USE_ASYNC_LDS 1
#  endif
#endif
#ifndef USE_ASYNC_LDS
#  define USE_ASYNC_LDS 0
#endif

// ---------------------------------------------------------------------------
// Types / helpers
// ---------------------------------------------------------------------------
typedef __attribute__((ext_vector_type(16))) __bf16 v16bf;
typedef __attribute__((ext_vector_type(8)))  float  v8f;
typedef __attribute__((ext_vector_type(4)))  int    v4i;

union FragAB { v16bf v; unsigned short h[16]; unsigned int u[8]; uint4 q[2]; };
union FragC  { v8f v; float f[8]; float4 f4[2]; };
union PackB  { unsigned short h[8]; uint4 q; };
union Q4H    { uint4 q; unsigned short h[8]; };

__device__ __forceinline__ float bf2f(unsigned short h) {
  union { unsigned int u; float f; } c; c.u = ((unsigned int)h) << 16; return c.f;
}
__device__ __forceinline__ unsigned short f2bf(float f) {
  union { float f; unsigned int u; } c; c.f = f;
  unsigned int u = c.u + 0x7FFFu + ((c.u >> 16) & 1u);   // RNE
  return (unsigned short)(u >> 16);
}
// A-fragment K offset, 16-bit 16x32 A (contiguous order: k = 16*hi + s)
__device__ __forceinline__ int klo_a(int v, int hi) {
  return (hi << 4) + (v << 1);
}

#if USE_ASYNC_LDS
__device__ __forceinline__ void stage_b128_async(const unsigned short* g,
                                                 unsigned short* l) {
  __builtin_amdgcn_global_load_async_to_lds_b128(
      (__attribute__((address_space(1))) v4i*)(void*)g,
      (__attribute__((address_space(3))) v4i*)(void*)l, 0, 0);
}
#endif

// ---------------------------------------------------------------------------
// f32 NCHW -> bf16 NHWC (strided channel placement for free concat)
// ---------------------------------------------------------------------------
__global__ void cvt_nchw_nhwc_kernel(const float* __restrict__ in,
                                     unsigned short* __restrict__ out,
                                     int pShift, int ostride, int ooff, int total) {
  int idx = blockIdx.x * blockDim.x + threadIdx.x;
  if (idx >= total) return;
  int c   = idx >> pShift;
  int pix = idx & ((1 << pShift) - 1);
  out[(size_t)pix * ostride + ooff + c] = f2bf(in[idx]);
}

// ---------------------------------------------------------------------------
// Weight repack: f32 OIHW -> bf16 [CoutPad][9*Cin], K tap-major: k = t*Cin+ci.
// Rows m >= Cout zero-padded (removes all guards from the GEMM staging).
// ---------------------------------------------------------------------------
__global__ void repack_w_kernel(const float* __restrict__ w,
                                unsigned short* __restrict__ out,
                                int Cout, int Cin, int CoutPad) {
  int Ktot  = Cin * 9;
  int total = CoutPad * Ktot;
  int idx = blockIdx.x * blockDim.x + threadIdx.x;
  if (idx >= total) return;
  int m  = idx / Ktot;
  int k  = idx - m * Ktot;
  int t  = k / Cin;
  int ci = k - t * Cin;
  unsigned short v = 0;
  if (m < Cout) v = f2bf(w[((size_t)m * Cin + ci) * 9 + t]);
  out[idx] = v;
}

// ---------------------------------------------------------------------------
// Bilinear 2x upsample, NHWC bf16, half-pixel centers, edge clamp, out scale
// ---------------------------------------------------------------------------
__global__ void up2_kernel(const unsigned short* __restrict__ in, int istride, int ioff,
                           unsigned short* __restrict__ out, int ostride, int ooff,
                           int winShift, float scale, int total) {
  int idx = blockIdx.x * blockDim.x + threadIdx.x;
  if (idx >= total) return;
  int Win = 1 << winShift;
  int Wo  = Win << 1;
  int c = idx & 63;
  int p = idx >> 6;
  int xo = p & (Wo - 1);
  int yo = p >> (winShift + 1);
  float yi = 0.5f * yo - 0.25f;
  float xi = 0.5f * xo - 0.25f;
  float y0f = floorf(yi), x0f = floorf(xi);
  int y0 = (int)y0f, x0 = (int)x0f;
  float wy = yi - y0f, wx = xi - x0f;
  int y0c = min(max(y0, 0), Win - 1), y1c = min(max(y0 + 1, 0), Win - 1);
  int x0c = min(max(x0, 0), Win - 1), x1c = min(max(x0 + 1, 0), Win - 1);
  const unsigned short* b = in + ioff + c;
  float v = (1.f - wy) * ((1.f - wx) * bf2f(b[(size_t)((y0c << winShift) + x0c) * istride])
                        +        wx  * bf2f(b[(size_t)((y0c << winShift) + x1c) * istride]))
          +        wy  * ((1.f - wx) * bf2f(b[(size_t)((y1c << winShift) + x0c) * istride])
                        +        wx  * bf2f(b[(size_t)((y1c << winShift) + x1c) * istride]));
  out[(size_t)p * ostride + ooff + c] = f2bf(v * scale);
}

// ---------------------------------------------------------------------------
// 3x3 conv (pad 1), implicit GEMM on v_wmma_f32_16x16x32_bf16.
// act NHWC bf16 [P][astride] (+aoff), wgt [CoutPad][9*Cin] tap-major,
// 2 M-tiles per wave (B reused), per-chunk single tap -> one bounds check
// and two b128 loads per B fragment. Weight tiles double-buffered in LDS
// (async global->LDS when available). Grid: x = P/64, y = CoutPad/32.
// ---------------------------------------------------------------------------
__global__ __launch_bounds__(128) void conv3x3_wmma_kernel(
    const unsigned short* __restrict__ act, int astride, int aoff,
    const unsigned short* __restrict__ wgt,
    const float* __restrict__ bias, int Cout,
    unsigned short* __restrict__ out_bf16,
    float* __restrict__ out_f32,
    int ostride, int ooff,
    int cinShift, int wShift, int H, int do_lrelu) {
  const int W     = 1 << wShift;
  const int Cmask = (1 << cinShift) - 1;
  const int Ktot  = 9 << cinShift;
  const int lane  = threadIdx.x & 31;
  const int wave  = threadIdx.x >> 5;
  const int hi    = lane >> 4;
  const int mn    = lane & 15;
  const int m0    = blockIdx.y * 32;
  const int n     = (blockIdx.x * 4 + wave) * 16 + mn;
  const int y     = n >> wShift;
  const int x     = n & (W - 1);

  __shared__ __align__(16) unsigned short As[2][32 * 32];

  const int si  = threadIdx.x << 3;            // 8 shorts per thread
  const int smm = si >> 5, skl = si & 31;
  const unsigned short* wrow = wgt + (size_t)(m0 + smm) * Ktot + skl;

  // Prologue: stage K-chunk 0
#if USE_ASYNC_LDS
  stage_b128_async(wrow, &As[0][si]);
  __builtin_amdgcn_s_wait_asynccnt(0);
#else
  *(uint4*)&As[0][si] = *(const uint4*)wrow;
#endif
  __syncthreads();

  FragC c0, c1;
#pragma unroll
  for (int r = 0; r < 8; ++r) { c0.f[r] = 0.f; c1.f[r] = 0.f; }

  int buf = 0;
  for (int kk = 0; kk < Ktot; kk += 32) {
    const bool more = (kk + 32) < Ktot;
    // Kick off next weight tile while this one is consumed
#if USE_ASYNC_LDS
    if (more) stage_b128_async(wrow + kk + 32, &As[buf ^ 1][si]);
#else
    uint4 wnext = make_uint4(0u, 0u, 0u, 0u);
    if (more) {
      wnext = *(const uint4*)(wrow + kk + 32);
      __builtin_prefetch(wrow + kk + 64, 0, 3);
    }
#endif

    const int kb = kk + (hi << 4);
    const int t  = kb >> cinShift;
    const int cb = kb & Cmask;
    const int ky = (t >= 6) ? 2 : ((t >= 3) ? 1 : 0);
    const int kx = t - 3 * ky;
    const int yy = y + ky - 1;
    const int xx = x + kx - 1;

    FragAB a0, a1, b;
    if (((unsigned)yy < (unsigned)H) & ((unsigned)xx < (unsigned)W)) {
      const unsigned short* src =
          act + (size_t)((yy << wShift) + xx) * astride + aoff + cb;
      b.q[0] = *(const uint4*)src;
      b.q[1] = *(const uint4*)(src + 8);
    } else {
      b.q[0] = make_uint4(0u, 0u, 0u, 0u);
      b.q[1] = make_uint4(0u, 0u, 0u, 0u);
    }
#pragma unroll
    for (int v = 0; v < 8; ++v) {
      const int ka = klo_a(v, hi);
      a0.u[v] = *(const unsigned int*)&As[buf][mn * 32 + ka];
      a1.u[v] = *(const unsigned int*)&As[buf][(16 + mn) * 32 + ka];
    }
    c0.v = __builtin_amdgcn_wmma_f32_16x16x32_bf16(false, a0.v, false, b.v,
                                                   (short)0, c0.v, false, false);
    c1.v = __builtin_amdgcn_wmma_f32_16x16x32_bf16(false, a1.v, false, b.v,
                                                   (short)0, c1.v, false, false);

#if USE_ASYNC_LDS
    if (more) __builtin_amdgcn_s_wait_asynccnt(0);
#else
    if (more) *(uint4*)&As[buf ^ 1][si] = wnext;
#endif
    __syncthreads();
    buf ^= 1;
  }

#pragma unroll
  for (int mt = 0; mt < 2; ++mt) {
    const FragC* cc = mt ? &c1 : &c0;
    const int mbase = m0 + mt * 16 + 8 * hi;
    FragC r; PackB pk;
#pragma unroll
    for (int j = 0; j < 8; ++j) {
      const int m = mbase + j;
      float f = cc->f[j] + ((m < Cout) ? bias[m] : 0.f);
      if (do_lrelu) f = (f >= 0.f) ? f : 0.1f * f;
      r.f[j] = f;
      pk.h[j] = f2bf(f);
    }
    const size_t ob = (size_t)n * ostride + ooff + mbase;
    if (out_bf16) *(uint4*)(out_bf16 + ob) = pk.q;
    if (out_f32) {
      *(float4*)(out_f32 + ob)     = r.f4[0];
      *(float4*)(out_f32 + ob + 4) = r.f4[1];
    }
  }
}

// ---------------------------------------------------------------------------
// Modulated deformable conv v2 aggregation on WMMA.
// xg NHWC bf16 (64 ch at xoff, row stride xstride), com NHWC f32 [P][224]
// (dy: g*9+t, dx: +72, mask-logit: +144), wgt [64][576] tap-major.
// Per 16-wide K block: one tap, 2 deformable groups -> 2 offset/sigmoid
// computations, each bilinear corner one b128 load covering 8 channels.
// B fragment feeds 4 WMMAs (Cout = 64). Double-buffered weight staging.
// ---------------------------------------------------------------------------
__global__ __launch_bounds__(128) void dcn_wmma_kernel(
    const unsigned short* __restrict__ xg, int xstride, int xoff,
    const float* __restrict__ com,
    const unsigned short* __restrict__ wgt,
    const float* __restrict__ bias,
    unsigned short* __restrict__ out_bf16, int ostride, int ooff,
    float* __restrict__ out_f32_nchw,
    int wShift, int H) {
  const int W    = 1 << wShift;
  const int lane = threadIdx.x & 31;
  const int wave = threadIdx.x >> 5;
  const int hi   = lane >> 4;
  const int mn   = lane & 15;
  const int n    = (blockIdx.x * 4 + wave) * 16 + mn;
  const int y    = n >> wShift;
  const int x    = n & (W - 1);

  __shared__ __align__(16) unsigned short As[2][64 * 32];
  const int si  = threadIdx.x << 4;            // 16 shorts per thread
  const int smm = si >> 5, skl = si & 31;
  const unsigned short* wrow = wgt + (size_t)smm * 576 + skl;
  const float* comp = com + (size_t)n * 224;

#if USE_ASYNC_LDS
  stage_b128_async(wrow, &As[0][si]);
  stage_b128_async(wrow + 8, &As[0][si + 8]);
  __builtin_amdgcn_s_wait_asynccnt(0);
#else
  *(uint4*)&As[0][si]     = *(const uint4*)wrow;
  *(uint4*)&As[0][si + 8] = *(const uint4*)(wrow + 8);
#endif
  __syncthreads();

  FragC c[4];
#pragma unroll
  for (int mt = 0; mt < 4; ++mt)
#pragma unroll
    for (int r = 0; r < 8; ++r) c[mt].f[r] = 0.f;

  int buf = 0;
  for (int kk = 0; kk < 576; kk += 32) {
    const bool more = (kk + 32) < 576;
#if USE_ASYNC_LDS
    if (more) {
      stage_b128_async(wrow + kk + 32, &As[buf ^ 1][si]);
      stage_b128_async(wrow + kk + 40, &As[buf ^ 1][si + 8]);
    }
#else
    uint4 wn0 = make_uint4(0u, 0u, 0u, 0u), wn1 = wn0;
    if (more) {
      wn0 = *(const uint4*)(wrow + kk + 32);
      wn1 = *(const uint4*)(wrow + kk + 40);
      __builtin_prefetch(wrow + kk + 64, 0, 3);
    }
#endif

    const int kb = kk + (hi << 4);
    const int t  = kb >> 6;
    const int cb = kb & 63;
    const int ky = (t >= 6) ? 2 : ((t >= 3) ? 1 : 0);
    const int kx = t - 3 * ky;

    FragAB b;
#pragma unroll
    for (int half = 0; half < 2; ++half) {
      const int g  = (cb >> 3) + half;
      const int ch = g * 9 + t;
      const float dy = comp[ch];
      const float dx = comp[72 + ch];
      const float mr = comp[144 + ch];
      const float mask = 1.f / (1.f + __expf(-mr));
      const float py = (float)(y - 1 + ky) + dy;
      const float px = (float)(x - 1 + kx) + dx;
      const float y0f = floorf(py), x0f = floorf(px);
      const int y0 = (int)y0f, x0 = (int)x0f;
      const float wy = py - y0f, wx = px - x0f;
      float s[8];
#pragma unroll
      for (int j = 0; j < 8; ++j) s[j] = 0.f;
      const unsigned short* chp = xg + xoff + g * 8;
#pragma unroll
      for (int cy = 0; cy < 2; ++cy) {
#pragma unroll
        for (int cx = 0; cx < 2; ++cx) {
          const int yy = y0 + cy, xx = x0 + cx;
          const float cw = (cy ? wy : 1.f - wy) * (cx ? wx : 1.f - wx);
          if (((unsigned)yy < (unsigned)H) & ((unsigned)xx < (unsigned)W)) {
            Q4H qv;
            qv.q = *(const uint4*)(chp + (size_t)((yy << wShift) + xx) * xstride);
#pragma unroll
            for (int j = 0; j < 8; ++j) s[j] += cw * bf2f(qv.h[j]);
          }
        }
      }
#pragma unroll
      for (int j = 0; j < 8; ++j) b.h[half * 8 + j] = f2bf(s[j] * mask);
    }

#pragma unroll
    for (int mt = 0; mt < 4; ++mt) {
      FragAB a;
#pragma unroll
      for (int v = 0; v < 8; ++v)
        a.u[v] = *(const unsigned int*)&As[buf][(mt * 16 + mn) * 32 + klo_a(v, hi)];
      c[mt].v = __builtin_amdgcn_wmma_f32_16x16x32_bf16(false, a.v, false, b.v,
                                                        (short)0, c[mt].v, false, false);
    }

#if USE_ASYNC_LDS
    if (more) __builtin_amdgcn_s_wait_asynccnt(0);
#else
    if (more) {
      *(uint4*)&As[buf ^ 1][si]     = wn0;
      *(uint4*)&As[buf ^ 1][si + 8] = wn1;
    }
#endif
    __syncthreads();
    buf ^= 1;
  }

  const int P = H << wShift;
#pragma unroll
  for (int mt = 0; mt < 4; ++mt) {
    const int mbase = mt * 16 + 8 * hi;
    FragC r; PackB pk;
#pragma unroll
    for (int j = 0; j < 8; ++j) {
      float f = c[mt].f[j] + bias[mbase + j];
      f = (f >= 0.f) ? f : 0.1f * f;          // dcn output always lrelu'd
      r.f[j] = f;
      pk.h[j] = f2bf(f);
    }
    if (out_bf16)
      *(uint4*)(out_bf16 + (size_t)n * ostride + ooff + mbase) = pk.q;
    if (out_f32_nchw) {
#pragma unroll
      for (int j = 0; j < 8; ++j)
        out_f32_nchw[(size_t)(mbase + j) * P + n] = r.f[j];   // NCHW final output
    }
  }
}

// ---------------------------------------------------------------------------
// Host orchestration
// ---------------------------------------------------------------------------
extern "C" void kernel_launch(void* const* d_in, const int* in_sizes, int n_in,
                              void* d_out, int out_size, void* d_ws, size_t ws_size,
                              hipStream_t stream) {
  (void)in_sizes; (void)n_in; (void)out_size; (void)ws_size;
  const int SH3 = 5, SH2 = 6, SH1 = 7, SH1r = 8;      // log2 widths
  const int S3 = 32, S2 = 64, S1 = 128, S1r = 256;
  const int P3 = S3 * S3, P2 = S2 * S2, P1 = S1 * S1, P1r = S1r * S1r;

  char* base = (char*)d_ws;
  size_t off = 0;
  auto alloc = [&](size_t elems, size_t esz) -> void* {
    void* p = base + off;
    off += ((elems * esz + 255) & ~(size_t)255);
    return p;
  };
  auto allocb = [&](size_t elems) -> unsigned short* {
    return (unsigned short*)alloc(elems, 2);
  };

  struct WD { int wi, bi, co, ci; };
  const WD wd[20] = {
    { 9, 10,  64, 128}, {11, 12,  64,  64}, {13, 14, 216,  64}, {15, 16,  64,  64}, // L3
    {17, 18,  64, 128}, {19, 20,  64, 128}, {21, 22,  64,  64}, {23, 24, 216,  64},
    {25, 26,  64,  64}, {27, 28,  64, 128},                                          // L2
    {29, 30,  64, 128}, {31, 32,  64, 128}, {33, 34,  64,  64}, {35, 36, 216,  64},
    {37, 38,  64,  64}, {39, 40,  64, 128},                                          // L1
    {41, 42,  64, 128}, {43, 44,  64,  64}, {45, 46, 216,  64}, {47, 48,  64,  64}, // cas
  };
  unsigned short* wb[20];
  const float* bias[20];
  int cop[20];
  for (int i = 0; i < 20; ++i) {
    cop[i]  = (wd[i].co + 31) & ~31;               // pad Cout to M-supertile
    wb[i]   = allocb((size_t)cop[i] * wd[i].ci * 9);
    bias[i] = (const float*)d_in[wd[i].bi];
  }

  // NHWC bf16 buffers; 128-wide buffers hold concat pairs in channel halves.
  unsigned short* catL3 = allocb((size_t)P3 * 128);  // [rp_nbr_l3 | rp_ref_l3]
  unsigned short* catL2 = allocb((size_t)P2 * 128);  // [rp_nbr_l2 | rp_ref_l2]
  unsigned short* catL1 = allocb((size_t)P1 * 128);  // [rp_nbr_l1 | rp_ref_l1] (off0 later = rp1f)
  unsigned short* xraw3 = allocb((size_t)P2 * 64);
  unsigned short* xraw2 = allocb((size_t)P1 * 64);
  unsigned short* xraw1 = allocb((size_t)P1r * 64);
  unsigned short* rp3oa = allocb((size_t)P3 * 64);
  unsigned short* rp3o  = allocb((size_t)P3 * 64);
  unsigned short* rp3f  = allocb((size_t)P3 * 64);
  unsigned short* raw3f = allocb((size_t)P2 * 64);
  unsigned short* catD  = allocb((size_t)P2 * 128);  // [rp2oa | o3up2]
  unsigned short* rp2ob = allocb((size_t)P2 * 64);
  unsigned short* rp2o  = allocb((size_t)P2 * 64);
  unsigned short* catE  = allocb((size_t)P2 * 128);  // [rp2f | rp3f_up]
  unsigned short* catF  = allocb((size_t)P1 * 128);  // [raw2f | raw3f_up]
  unsigned short* rp2f2 = allocb((size_t)P2 * 64);
  unsigned short* raw2f2= allocb((size_t)P1 * 64);
  unsigned short* catG  = allocb((size_t)P1 * 128);  // [rp1oa | o2up2]
  unsigned short* rp1ob = allocb((size_t)P1 * 64);
  unsigned short* rp1o  = allocb((size_t)P1 * 64);
  unsigned short* o1up2 = allocb((size_t)P1r * 64);
  unsigned short* catH  = allocb((size_t)P1 * 128);  // [rp1f | rp2f_up]
  unsigned short* catI  = allocb((size_t)P1r * 128); // [raw1f | raw2f_up]
  unsigned short* raw1f2= allocb((size_t)P1r * 64);
  unsigned short* coa   = allocb((size_t)P1 * 64);
  unsigned short* co    = allocb((size_t)P1 * 64);
  unsigned short* coup2 = allocb((size_t)P1r * 64);
  float* comf = (float*)alloc((size_t)P1r * 224, 4); // reused offset/mask (stride 224)

  auto cvt = [&](int ii, unsigned short* d, int ostride, int ooff, int pShift) {
    int total = 64 << pShift;
    cvt_nchw_nhwc_kernel<<<dim3((unsigned)((total + 255) / 256)), 256, 0, stream>>>(
        (const float*)d_in[ii], d, pShift, ostride, ooff, total);
  };
  for (int i = 0; i < 20; ++i) {
    int total = cop[i] * wd[i].ci * 9;
    repack_w_kernel<<<dim3((unsigned)((total + 255) / 256)), 256, 0, stream>>>(
        (const float*)d_in[wd[i].wi], wb[i], wd[i].co, wd[i].ci, cop[i]);
  }
  cvt(5, catL3, 128, 0, 10); cvt(8, catL3, 128, 64, 10);
  cvt(4, catL2, 128, 0, 12); cvt(7, catL2, 128, 64, 12);
  cvt(3, catL1, 128, 0, 14); cvt(6, catL1, 128, 64, 14);
  cvt(2, xraw3, 64, 0, 12);  cvt(1, xraw2, 64, 0, 14);  cvt(0, xraw1, 64, 0, 16);

  auto conv = [&](const unsigned short* a, int astr, int aoff, int cinSh, int wi,
                  int wSh, unsigned short* ob, float* of, int ostr, int ooff, int lr) {
    int S = 1 << wSh;
    dim3 g((unsigned)(S * S / 64), (unsigned)(cop[wi] / 32));
    conv3x3_wmma_kernel<<<g, 128, 0, stream>>>(a, astr, aoff, wb[wi], bias[wi],
                                               wd[wi].co, ob, of, ostr, ooff,
                                               cinSh, wSh, S, lr);
  };
  auto dcnk = [&](const unsigned short* x, int xstr, int xoff, int wi, int wSh,
                  unsigned short* ob, int ostr, int ooff, float* of) {
    int S = 1 << wSh;
    dim3 g((unsigned)(S * S / 64));
    dcn_wmma_kernel<<<g, 128, 0, stream>>>(x, xstr, xoff, comf, wb[wi], bias[wi],
                                           ob, ostr, ooff, of, wSh, S);
  };
  auto up2 = [&](const unsigned short* s, int istr, int ioff,
                 unsigned short* d, int ostr, int ooff, int winSh, float sc) {
    int total = 64 << (2 * winSh + 2);
    up2_kernel<<<dim3((unsigned)((total + 255) / 256)), 256, 0, stream>>>(
        s, istr, ioff, d, ostr, ooff, winSh, sc, total);
  };

  float* out = (float*)d_out;

  // ---- L3 ----
  conv(catL3, 128, 0, 7, 0, SH3, rp3oa, nullptr, 64, 0, 1);   // L3_oc1
  conv(rp3oa,  64, 0, 6, 1, SH3, rp3o,  nullptr, 64, 0, 1);   // L3_oc2
  up2(rp3o, 64, 0, catD, 128, 64, SH3, 2.0f);                 // o3up*2
  conv(rp3o,   64, 0, 6, 2, SH3, nullptr, comf, 224, 0, 0);   // offsets rp
  dcnk(catL3, 128, 0, 3, SH3, rp3f, 64, 0, nullptr);
  conv(catD, 128, 64, 6, 2, SH2, nullptr, comf, 224, 0, 0);   // offsets raw (o3up2)
  dcnk(xraw3,  64, 0, 3, SH2, raw3f, 64, 0, nullptr);
  // ---- L2 ----
  conv(catL2, 128, 0, 7, 4, SH2, catD, nullptr, 128, 0, 1);   // L2_oc1 -> rp2oa
  conv(catD,  128, 0, 7, 5, SH2, rp2ob, nullptr, 64, 0, 1);   // L2_oc2(cat)
  conv(rp2ob,  64, 0, 6, 6, SH2, rp2o,  nullptr, 64, 0, 1);   // L2_oc3
  up2(rp2o, 64, 0, catG, 128, 64, SH2, 2.0f);                 // o2up*2
  conv(rp2o,   64, 0, 6, 7, SH2, nullptr, comf, 224, 0, 0);
  dcnk(catL2, 128, 0, 8, SH2, catE, 128, 0, nullptr);         // rp2f
  conv(catG, 128, 64, 6, 7, SH1, nullptr, comf, 224, 0, 0);   // from o2up2
  dcnk(xraw2,  64, 0, 8, SH1, catF, 128, 0, nullptr);         // raw2f
  up2(rp3f, 64, 0, catE, 128, 64, SH3, 1.0f);
  up2(raw3f, 64, 0, catF, 128, 64, SH2, 1.0f);
  conv(catE, 128, 0, 7, 9, SH2, rp2f2,  nullptr, 64, 0, 1);   // L2_fc
  conv(catF, 128, 0, 7, 9, SH1, raw2f2, nullptr, 64, 0, 1);   // L2_fc
  // ---- L1 ----
  conv(catL1, 128, 0, 7, 10, SH1, catG, nullptr, 128, 0, 1);  // L1_oc1 -> rp1oa
  conv(catG,  128, 0, 7, 11, SH1, rp1ob, nullptr, 64, 0, 1);  // L1_oc2(cat)
  conv(rp1ob,  64, 0, 6, 12, SH1, rp1o, nullptr, 64, 0, 1);   // L1_oc3
  up2(rp1o, 64, 0, o1up2, 64, 0, SH1, 2.0f);                  // o1up*2
  conv(rp1o,   64, 0, 6, 13, SH1, nullptr, comf, 224, 0, 0);
  dcnk(catL1, 128, 0, 14, SH1, catH, 128, 0, nullptr);        // rp1f
  conv(o1up2,  64, 0, 6, 13, SH1r, nullptr, comf, 224, 0, 0);
  dcnk(xraw1,  64, 0, 14, SH1r, catI, 128, 0, nullptr);       // raw1f
  up2(rp2f2, 64, 0, catH, 128, 64, SH2, 1.0f);
  up2(raw2f2, 64, 0, catI, 128, 64, SH1, 1.0f);
  conv(catH, 128, 0, 7, 15, SH1, catL1, nullptr, 128, 0, 1);  // L1_fc -> rp1f (dead nbr slot)
  conv(catI, 128, 0, 7, 15, SH1r, raw1f2, nullptr, 64, 0, 1); // L1_fc
  // ---- cascade ----
  conv(catL1, 128, 0, 7, 16, SH1, coa, nullptr, 64, 0, 1);    // cas_oc1(cat(rp1f,ref))
  conv(coa,    64, 0, 6, 17, SH1, co,  nullptr, 64, 0, 1);    // cas_oc2
  up2(co, 64, 0, coup2, 64, 0, SH1, 2.0f);                    // coup*2
  conv(co,     64, 0, 6, 18, SH1, nullptr, comf, 224, 0, 0);
  dcnk(catL1, 128, 0, 19, SH1, nullptr, 0, 0, out + (size_t)64 * P1r); // final rp1f f32
  conv(coup2,  64, 0, 6, 18, SH1r, nullptr, comf, 224, 0, 0);
  dcnk(raw1f2, 64, 0, 19, SH1r, nullptr, 0, 0, out);                   // final raw1f f32
}
